// DNN_19507741458922
// MI455X (gfx1250) — compile-verified, matrix operand order checked by reading, not run on hardware
//
#include <hip/hip_runtime.h>

// ---------------------------------------------------------------------------
// Problem constants (reference): BS=64, D_NUM=2, SLEN=50, DIM=64
// N = 6400 rows, NUM_EMB = 512, outputs: quantized | Z | X, each 409600 f32.
// ---------------------------------------------------------------------------
#define NROWS     6400
#define DIM       64
#define NUM_EMB   512
#define OUT_BLK   409600      // 6400*64
#define SAMP_ROWS 100         // dn*sl rows per sample (mask block size)

typedef __attribute__((ext_vector_type(16))) _Float16 v16h;
typedef __attribute__((ext_vector_type(8)))  float    v8f;

// ---------------------------------------------------------------------------
// Fragment-order layout: a 16x32 f16 tile is 512 contiguous halves arranged
// exactly as the wave32 WMMA VGPR layout (ISA 7.12.2):
//   lane = ((k>>3)&1)*16 + rc ,  elem = (k&7) + ((k>>4)<<3)
// so a fragment load is one contiguous v16h (= 2x b128) per lane.
// ---------------------------------------------------------------------------
__device__ __forceinline__ int frag_off(int rc, int kk) {
  return ((((kk >> 3) & 1) << 4) + rc) * 16 + (kk & 7) + ((kk >> 4) << 3);
}

__device__ __forceinline__ v16h load_frag(const _Float16* tile) {
  return *(const v16h*)(tile + (threadIdx.x & 31) * 16);
}

// A-role workspace layout (q_cs, k_cs): tiles of 16 rows x 32 k
__device__ __forceinline__ int ws_fragA_index(int row, int col) {
  return ((row >> 4) * 2 + (col >> 5)) * 512 + frag_off(row & 15, col & 31);
}
// B-role workspace layout (v_cs): K = row (N index), N = col (dim index)
__device__ __forceinline__ int ws_fragB_index(int row, int col) {
  return ((row >> 6) * 8 + (col >> 4) * 2 + ((row & 63) >> 5)) * 512 +
         frag_off(col & 15, row & 31);
}

__device__ __forceinline__ v8f wmma_f16(v16h a, v16h b, v8f c) {
  return __builtin_amdgcn_wmma_f32_16x16x32_f16(
      /*neg_a=*/false, a, /*neg_b=*/false, b,
      /*c_mod=*/(short)0, c, /*reuse_a=*/false, /*reuse_b=*/false);
}

// ---------------------------------------------------------------------------
// Kernel 1: vector quantization. argmin_j (||c_j||^2 - 2 f.c_j), gather row.
// Codebook streamed through LDS in 128-row chunks; fp32 kept end-to-end so
// the argmin matches the reference. float4 dot loop -> ds_load_b128.
// 25 blocks x 256 threads.
// ---------------------------------------------------------------------------
__global__ void __launch_bounds__(256)
quant_kernel(const float* __restrict__ x, const float* __restrict__ cb,
             float* __restrict__ outq) {
  __shared__ __align__(16) float Cs[128 * DIM];
  __shared__ float Cn[128];
  const int tid = threadIdx.x;
  const int row = blockIdx.x * 256 + tid;   // 25*256 == 6400 exactly

  float4 f4[DIM / 4];
#pragma unroll
  for (int i = 0; i < DIM / 4; ++i)
    f4[i] = *(const float4*)(x + row * DIM + i * 4);

  float best  = 3.4e38f;
  int   bestj = 0;

  for (int ch = 0; ch < NUM_EMB / 128; ++ch) {
    __syncthreads();
    {
      const float4* src = (const float4*)(cb + ch * 128 * DIM);
      float4*       dst = (float4*)Cs;
      for (int i = tid; i < (128 * DIM) / 4; i += 256) dst[i] = src[i];
    }
    __syncthreads();
    if (tid < 128) {
      float s = 0.f;
#pragma unroll
      for (int i = 0; i < DIM; ++i) { float v = Cs[tid * DIM + i]; s += v * v; }
      Cn[tid] = s;
    }
    __syncthreads();
    for (int j = 0; j < 128; ++j) {
      const float4* c4 = (const float4*)(Cs + j * DIM);
      float dot = 0.f;
#pragma unroll
      for (int i = 0; i < DIM / 4; ++i) {
        const float4 cv = c4[i];
        dot = fmaf(f4[i].x, cv.x, dot);
        dot = fmaf(f4[i].y, cv.y, dot);
        dot = fmaf(f4[i].z, cv.z, dot);
        dot = fmaf(f4[i].w, cv.w, dot);
      }
      const float sc = Cn[j] - 2.0f * dot;
      if (sc < best) { best = sc; bestj = ch * 128 + j; }  // strict <: first-min
    }
  }
#pragma unroll
  for (int i = 0; i < DIM / 4; ++i)
    *(float4*)(outq + row * DIM + i * 4) = *(const float4*)(cb + bestj * DIM + i * 4);
}

// ---------------------------------------------------------------------------
// Kernel 2: six 64x64 projections. x and W staged in fragment-order f16 LDS,
// WMMA with bias in the accumulator. IS outputs row-major; CS outputs are
// written to workspace PRE-SWIZZLED for the flash kernel. The projection loop
// is fully unrolled so the per-output layout choice constant-folds (no
// per-element branches around stores).
// 100 blocks x 128 threads; block owns 64 rows; wave w owns 16 rows.
// ---------------------------------------------------------------------------
struct ProjArgs {
  const float* W[6];
  const float* b[6];
  _Float16*    out[6];
};

__global__ void __launch_bounds__(128)
proj_kernel(const float* __restrict__ x, ProjArgs pa) {
  __shared__ __align__(32) _Float16 Af[8 * 512];   // 4 row strips x 2 k tiles
  __shared__ __align__(32) _Float16 Bf[8 * 512];   // 4 col tiles  x 2 k tiles
  const int tid  = threadIdx.x;
  const int wave = tid >> 5;
  const int lane = tid & 31;
  const int col0 = lane & 15;
  const int rb   = (lane >> 4) * 8;
  const int rowbase = blockIdx.x * 64;

  // stage x (fp32 -> f16) in fragment order; pairs along k are contiguous
  for (int i = tid; i < 2048; i += 128) {
    const int r = i >> 5, c = (i & 31) * 2;
    const float2 xv = *(const float2*)(x + (rowbase + r) * DIM + c);
    const int off = (r >> 4) * 1024 + (c >> 5) * 512 + frag_off(r & 15, c & 31);
    Af[off]     = (_Float16)xv.x;
    Af[off + 1] = (_Float16)xv.y;
  }
  __syncthreads();

  const v16h a0 = load_frag(Af + wave * 1024);
  const v16h a1 = load_frag(Af + wave * 1024 + 512);

#pragma unroll
  for (int p = 0; p < 6; ++p) {
    // stage W (row-major KxN) in B fragment order; pairs along k contiguous
    for (int i = tid; i < 2048; i += 128) {
      const int n = i & 63, k = (i >> 6) * 2;
      const float w0 = pa.W[p][k * DIM + n];
      const float w1 = pa.W[p][(k + 1) * DIM + n];
      const int off = ((n >> 4) * 2 + (k >> 5)) * 512 + frag_off(n & 15, k & 31);
      Bf[off]     = (_Float16)w0;
      Bf[off + 1] = (_Float16)w1;
    }
    __syncthreads();

#pragma unroll
    for (int ct = 0; ct < 4; ++ct) {
      const float bias = pa.b[p][ct * 16 + col0];
      v8f c;
#pragma unroll
      for (int i = 0; i < 8; ++i) c[i] = bias;         // bias broadcast per col
      const v16h b0 = load_frag(Bf + (ct * 2) * 512);
      const v16h b1 = load_frag(Bf + (ct * 2) * 512 + 512);
      c = wmma_f16(a0, b0, c);
      c = wmma_f16(a1, b1, c);
#pragma unroll
      for (int v = 0; v < 8; ++v) {
        const int r  = rowbase + wave * 16 + rb + v;
        const int cc = ct * 16 + col0;
        const _Float16 h = (_Float16)c[v];
        if (p < 3)       pa.out[p][r * DIM + cc]          = h;  // row-major (IS)
        else if (p < 5)  pa.out[p][ws_fragA_index(r, cc)] = h;  // q_cs / k_cs
        else             pa.out[p][ws_fragB_index(r, cc)] = h;  // v_cs
      }
    }
    __syncthreads();
  }
}

// ---------------------------------------------------------------------------
// Kernel 3: intra-sample attention, one block per (b,d); 50 padded to 64.
// K/Q/V staged in fragment-order LDS (zero padded); S = K Q^T via WMMA,
// exact softmax over the 50 real columns, P written in fragment order,
// Z = P V via WMMA.
// ---------------------------------------------------------------------------
__global__ void __launch_bounds__(128)
is_attn_kernel(const _Float16* __restrict__ q, const _Float16* __restrict__ k,
               const _Float16* __restrict__ vmat, float* __restrict__ Z) {
  __shared__ __align__(32) _Float16 Kf[8 * 512];
  __shared__ __align__(32) _Float16 Qf[8 * 512];
  __shared__ __align__(32) _Float16 Vf[8 * 512];
  __shared__ __align__(32) _Float16 Pf[8 * 512];
  __shared__ float Sf[64 * 64];

  const int tid  = threadIdx.x;
  const int wave = tid >> 5;
  const int lane = tid & 31;
  const int col0 = lane & 15;
  const int rb   = (lane >> 4) * 8;
  const int rbase = blockIdx.x * 50;   // first flat row of this (b,d)

  // ---- stage K, Q (pairs along k contiguous; zero pad rows >= 50) ----
  for (int i = tid; i < 2048; i += 128) {
    const int r = i >> 5, c = (i & 31) * 2;
    unsigned kv = 0u, qv = 0u;
    if (r < 50) {
      kv = *(const unsigned*)(k + (rbase + r) * DIM + c);
      qv = *(const unsigned*)(q + (rbase + r) * DIM + c);
    }
    const int off = (r >> 4) * 1024 + (c >> 5) * 512 + frag_off(r & 15, c & 31);
    *(unsigned*)(Kf + off) = kv;
    *(unsigned*)(Qf + off) = qv;
  }
  // ---- stage V in B-fragment order (K dim = key row; pairs along r) ----
  for (int i = tid; i < 2048; i += 128) {
    const int c = i & 63, r = (i >> 6) * 2;
    _Float16 v0 = (_Float16)0.f, v1 = (_Float16)0.f;
    if (r < 50)     v0 = vmat[(rbase + r) * DIM + c];
    if (r + 1 < 50) v1 = vmat[(rbase + r + 1) * DIM + c];
    const int off = ((c >> 4) * 2 + (r >> 5)) * 512 + frag_off(c & 15, r & 31);
    Vf[off]     = v0;
    Vf[off + 1] = v1;
  }
  // ---- zero P (padded region must stay zero) ----
  {
    uint4* pz = (uint4*)Pf;
    for (int i = tid; i < 512; i += 128) pz[i] = make_uint4(0, 0, 0, 0);
  }
  __syncthreads();

  // ---- S = K Q^T, strip of 16 rows per wave ----
  {
    const v16h ka0 = load_frag(Kf + wave * 1024);
    const v16h ka1 = load_frag(Kf + wave * 1024 + 512);
#pragma unroll
    for (int ct = 0; ct < 4; ++ct) {
      v8f c;
#pragma unroll
      for (int i = 0; i < 8; ++i) c[i] = 0.f;
      const v16h b0 = load_frag(Qf + (ct * 2) * 512);        // B[k][n]=Q[n][k]
      const v16h b1 = load_frag(Qf + (ct * 2) * 512 + 512);
      c = wmma_f16(ka0, b0, c);
      c = wmma_f16(ka1, b1, c);
#pragma unroll
      for (int v = 0; v < 8; ++v)
        Sf[(wave * 16 + rb + v) * 64 + ct * 16 + col0] = c[v];
    }
  }
  __syncthreads();

  // ---- softmax over the 50 real cols; write normalized P in frag order ----
  if (tid < 50) {
    const int r = tid;
    float mx = -3.4e38f;
    for (int t = 0; t < 50; ++t) mx = fmaxf(mx, Sf[r * 64 + t]);
    float s = 0.f;
    for (int t = 0; t < 50; ++t) s += __expf(Sf[r * 64 + t] - mx);
    const float inv = 1.0f / s;
    for (int t = 0; t < 50; ++t) {
      const float pv = __expf(Sf[r * 64 + t] - mx) * inv;
      Pf[(r >> 4) * 1024 + (t >> 5) * 512 + frag_off(r & 15, t & 31)] =
          (_Float16)pv;
    }
  }
  __syncthreads();

  // ---- Z = P V ----
  {
    const v16h pa0 = load_frag(Pf + wave * 1024);
    const v16h pa1 = load_frag(Pf + wave * 1024 + 512);
#pragma unroll
    for (int ct = 0; ct < 4; ++ct) {
      v8f c;
#pragma unroll
      for (int i = 0; i < 8; ++i) c[i] = 0.f;
      const v16h b0 = load_frag(Vf + (ct * 2) * 512);
      const v16h b1 = load_frag(Vf + (ct * 2) * 512 + 512);
      c = wmma_f16(pa0, b0, c);
      c = wmma_f16(pa1, b1, c);
#pragma unroll
      for (int v = 0; v < 8; ++v) {
        const int s = wave * 16 + rb + v;
        if (s < 50)
          Z[(rbase + s) * DIM + ct * 16 + col0] = c[v];
      }
    }
  }
}

// ---------------------------------------------------------------------------
// Kernel 4: cross-sample attention, flash style, barrier-free.
// Operands arrive pre-swizzled from the projection kernel, so every fragment
// is a contiguous global b128 load (shared tiles hit WGP$/L2). Only the
// P transpose (D-layout -> A-layout) bounces through per-wave LDS.
// scores = kc qc^T, own-sample 100x100 block masked to -1e30, X = P vc.
// 100 blocks x 128 threads; wave strip = 16 rows; 100 chunks of 64 keys.
// ---------------------------------------------------------------------------
__global__ void __launch_bounds__(128)
cs_attn_kernel(const _Float16* __restrict__ qc, const _Float16* __restrict__ kc,
               const _Float16* __restrict__ vc, float* __restrict__ X) {
  __shared__ __align__(32) _Float16 Pf[4 * 1024];   // per-wave P staging

  const int tid  = threadIdx.x;
  const int wave = tid >> 5;
  const int lane = tid & 31;
  const int col0 = lane & 15;
  const int rb   = (lane >> 4) * 8;
  const int row0 = blockIdx.x * 64 + wave * 16;  // this wave's rows (kc side)
  _Float16* Pw   = Pf + wave * 1024;

  // kc strip A-fragments: contiguous global loads, invariant across chunks
  const v16h a0 = load_frag(kc + (row0 >> 4) * 1024);
  const v16h a1 = load_frag(kc + (row0 >> 4) * 1024 + 512);

  v8f   acc[4];
  float m8[8], l8[8];
#pragma unroll
  for (int ct = 0; ct < 4; ++ct)
#pragma unroll
    for (int i = 0; i < 8; ++i) acc[ct][i] = 0.f;
#pragma unroll
  for (int i = 0; i < 8; ++i) { m8[i] = -3.4e38f; l8[i] = 0.f; }

  for (int j = 0; j < NROWS / 64; ++j) {
    const _Float16* Qj = qc + j * 4096;
    const _Float16* Vj = vc + j * 4096;
    if (j + 1 < NROWS / 64) {          // prefetch next 8KB key/value chunks
      __builtin_prefetch(qc + (j + 1) * 4096 + lane * 128, 0, 1);
      __builtin_prefetch(vc + (j + 1) * 4096 + lane * 128, 0, 1);
    }

    // ---- scores strip (16x64) ----
    v8f s[4];
#pragma unroll
    for (int ct = 0; ct < 4; ++ct) {
      v8f c;
#pragma unroll
      for (int i = 0; i < 8; ++i) c[i] = 0.f;
      const v16h b0 = load_frag(Qj + (ct * 2) * 512);   // B[k][n]=qc[n][k]
      const v16h b1 = load_frag(Qj + (ct * 2) * 512 + 512);
      c = wmma_f16(a0, b0, c);
      c = wmma_f16(a1, b1, c);
      // block-diagonal mask: same sample (100-row blocks) -> -1e30
#pragma unroll
      for (int v = 0; v < 8; ++v) {
        const int r  = row0 + rb + v;
        const int cg = j * 64 + ct * 16 + col0;
        if ((r / SAMP_ROWS) == (cg / SAMP_ROWS)) c[v] = -1e30f;
      }
      s[ct] = c;
    }

    // ---- online softmax update (rows live in 16-lane groups) ----
#pragma unroll
    for (int v = 0; v < 8; ++v) {
      float tm = fmaxf(fmaxf(s[0][v], s[1][v]), fmaxf(s[2][v], s[3][v]));
#pragma unroll
      for (int d = 1; d < 16; d <<= 1) tm = fmaxf(tm, __shfl_xor(tm, d, 32));
      const float mnew  = fmaxf(m8[v], tm);
      const float scale = __expf(m8[v] - mnew);
      float rs = 0.f;
#pragma unroll
      for (int ct = 0; ct < 4; ++ct) {
        const float p = __expf(s[ct][v] - mnew);
        s[ct][v] = p;
        rs += p;
      }
#pragma unroll
      for (int d = 1; d < 16; d <<= 1) rs += __shfl_xor(rs, d, 32);
      l8[v] = l8[v] * scale + rs;
      m8[v] = mnew;
#pragma unroll
      for (int ct = 0; ct < 4; ++ct) acc[ct][v] *= scale;
    }

    // ---- P: D-layout regs -> fragment-order LDS (per-wave, no barrier) ----
#pragma unroll
    for (int ct = 0; ct < 4; ++ct) {
      const int cc  = ct * 16 + col0;
      const int ks  = (cc >> 5) * 512;
#pragma unroll
      for (int v = 0; v < 8; ++v)
        Pw[ks + frag_off(rb + v, cc & 31)] = (_Float16)s[ct][v];
    }
    const v16h pa0 = load_frag(Pw);
    const v16h pa1 = load_frag(Pw + 512);

    // ---- acc += P V ----
#pragma unroll
    for (int ct = 0; ct < 4; ++ct) {
      const v16h b0 = load_frag(Vj + (ct * 2) * 512);
      const v16h b1 = load_frag(Vj + (ct * 2) * 512 + 512);
      acc[ct] = wmma_f16(pa0, b0, acc[ct]);
      acc[ct] = wmma_f16(pa1, b1, acc[ct]);
    }
  }

  // ---- epilogue: X = acc / l ----
#pragma unroll
  for (int ct = 0; ct < 4; ++ct) {
#pragma unroll
    for (int v = 0; v < 8; ++v) {
      const int r = row0 + rb + v;
      X[r * DIM + ct * 16 + col0] = acc[ct][v] / l8[v];
    }
  }
}

// ---------------------------------------------------------------------------
// Launcher
// ---------------------------------------------------------------------------
extern "C" void kernel_launch(void* const* d_in, const int* in_sizes, int n_in,
                              void* d_out, int out_size, void* d_ws, size_t ws_size,
                              hipStream_t stream) {
  const float* x     = (const float*)d_in[0];
  const float* cb    = (const float*)d_in[1];
  const float* Wq_is = (const float*)d_in[2];  const float* bq_is = (const float*)d_in[3];
  const float* Wk_is = (const float*)d_in[4];  const float* bk_is = (const float*)d_in[5];
  const float* Wv_is = (const float*)d_in[6];  const float* bv_is = (const float*)d_in[7];
  const float* Wq_cs = (const float*)d_in[8];  const float* bq_cs = (const float*)d_in[9];
  const float* Wk_cs = (const float*)d_in[10]; const float* bk_cs = (const float*)d_in[11];
  const float* Wv_cs = (const float*)d_in[12]; const float* bv_cs = (const float*)d_in[13];

  float* outq = (float*)d_out;
  float* outZ = outq + OUT_BLK;
  float* outX = outq + 2 * OUT_BLK;

  _Float16* ws   = (_Float16*)d_ws;     // 6 x 409600 f16 = ~4.7 MB scratch
  _Float16* q_is = ws + 0 * OUT_BLK;    // row-major
  _Float16* k_is = ws + 1 * OUT_BLK;    // row-major
  _Float16* v_is = ws + 2 * OUT_BLK;    // row-major
  _Float16* q_cs = ws + 3 * OUT_BLK;    // A-fragment chunk order
  _Float16* k_cs = ws + 4 * OUT_BLK;    // A-fragment chunk order
  _Float16* v_cs = ws + 5 * OUT_BLK;    // B-fragment chunk order

  quant_kernel<<<NROWS / 256, 256, 0, stream>>>(x, cb, outq);

  ProjArgs pa;
  pa.W[0] = Wq_is; pa.b[0] = bq_is; pa.out[0] = q_is;
  pa.W[1] = Wk_is; pa.b[1] = bk_is; pa.out[1] = k_is;
  pa.W[2] = Wv_is; pa.b[2] = bv_is; pa.out[2] = v_is;
  pa.W[3] = Wq_cs; pa.b[3] = bq_cs; pa.out[3] = q_cs;
  pa.W[4] = Wk_cs; pa.b[4] = bk_cs; pa.out[4] = k_cs;
  pa.W[5] = Wv_cs; pa.b[5] = bv_cs; pa.out[5] = v_cs;
  proj_kernel<<<NROWS / 64, 128, 0, stream>>>(x, pa);

  is_attn_kernel<<<128, 128, 0, stream>>>(q_is, k_is, v_is, outZ);
  cs_attn_kernel<<<NROWS / 64, 128, 0, stream>>>(q_cs, k_cs, v_cs, outX);
}
